// AttentionBlock_63591285784763
// MI455X (gfx1250) — matrix operand order repaired and checked by reading, not compiled, not run on hardware
//
#include <hip/hip_runtime.h>

typedef __attribute__((ext_vector_type(16))) __bf16 v16bf;
typedef __attribute__((ext_vector_type(4)))  __bf16 v4bf;
typedef __attribute__((ext_vector_type(8)))  float  v8f;

union FragA { v16bf v; uint4 u[2]; };

#define WMMA_BF16(a, b, c) \
  __builtin_amdgcn_wmma_f32_16x16x32_bf16(false, (a), false, (b), (short)0, (c), false, false)

// DPP16 lane exchange (VALU-only, no LDS): ctrl is an immediate DPP control.
#define DPP_F32(x, ctrl) \
  __int_as_float(__builtin_amdgcn_update_dpp(0, __float_as_int(x), (ctrl), 0xF, 0xF, true))

// 16-lane butterfly reductions (lanes 0-15 and 16-31 reduce independently;
// DPP16 "row" == 16 lanes). quad_perm(1,0,3,2)=0xB1 (xor1),
// quad_perm(2,3,0,1)=0x4E (xor2), row_half_mirror=0x141 (crosses quads),
// row_mirror=0x140 (crosses 8-lane halves).
__device__ __forceinline__ float red16_max(float x) {
  x = fmaxf(x, DPP_F32(x, 0xB1));
  x = fmaxf(x, DPP_F32(x, 0x4E));
  x = fmaxf(x, DPP_F32(x, 0x141));
  x = fmaxf(x, DPP_F32(x, 0x140));
  return x;
}
__device__ __forceinline__ float red16_sum(float x) {
  x += DPP_F32(x, 0xB1);
  x += DPP_F32(x, 0x4E);
  x += DPP_F32(x, 0x141);
  x += DPP_F32(x, 0x140);
  return x;
}

static constexpr int Bc = 4, Sc = 2048, Ec = 1024, Hc = 16, Dc = 64;
static constexpr size_t QK_ELEMS = (size_t)Bc * Hc * Sc * Dc;   // 8,388,608

// ---------------------------------------------------------------------------
// Phase 0a: fp32 [B,S,E] -> bf16 head-split [B*H, S, D], optional scale
// ---------------------------------------------------------------------------
__global__ __launch_bounds__(256) void cvt_split_kernel(const float* __restrict__ x,
                                                        __bf16* __restrict__ y,
                                                        float scale) {
  const int idx = blockIdx.x * 256 + threadIdx.x;   // 2,097,152 threads, 4 elems each
  const int e = idx * 4;
  const int d  = e & 63;
  const int s  = (e >> 6) & 2047;
  const int bh = e >> 17;
  const int b = bh >> 4, h = bh & 15;
  const float4 f = *(const float4*)(x + ((size_t)(b * Sc + s)) * Ec + h * Dc + d);
  v4bf o;
  o.x = (__bf16)(f.x * scale); o.y = (__bf16)(f.y * scale);
  o.z = (__bf16)(f.z * scale); o.w = (__bf16)(f.w * scale);
  *(v4bf*)(y + e) = o;
}

// ---------------------------------------------------------------------------
// Phase 0b: v fp32 [B,S,E] -> bf16 transposed [B*H, D, S] (LDS-tiled transpose)
// ---------------------------------------------------------------------------
__global__ __launch_bounds__(256) void vtrans_kernel(const float* __restrict__ v,
                                                     __bf16* __restrict__ vt) {
  __shared__ __bf16 tile[32][33];
  const int bh = blockIdx.z, b = bh >> 4, h = bh & 15;
  const int s0 = blockIdx.x * 32, d0 = blockIdx.y * 32;
  const int tx = threadIdx.x, ty = threadIdx.y;       // 32 x 8
  #pragma unroll
  for (int i = ty; i < 32; i += 8)
    tile[i][tx] = (__bf16)v[((size_t)(b * Sc + s0 + i)) * Ec + h * Dc + d0 + tx];
  __syncthreads();
  #pragma unroll
  for (int i = ty; i < 32; i += 8)
    vt[(((size_t)bh * Dc) + d0 + i) * Sc + s0 + tx] = tile[tx][i];
}

// ---------------------------------------------------------------------------
// Phase 0c: w_out fp32 [E,E] -> bf16 [E,E] (row n, contiguous e = WMMA B layout)
// ---------------------------------------------------------------------------
__global__ __launch_bounds__(256) void cvt_w_kernel(const float* __restrict__ x,
                                                    __bf16* __restrict__ y) {
  const int e = (blockIdx.x * 256 + threadIdx.x) * 4;
  const float4 f = *(const float4*)(x + e);
  v4bf o;
  o.x = (__bf16)f.x; o.y = (__bf16)f.y; o.z = (__bf16)f.z; o.w = (__bf16)f.w;
  *(v4bf*)(y + e) = o;
}

// ---------------------------------------------------------------------------
// Phase 1: flash attention. 8 waves/WG, 16 query rows per wave, 64-key tiles,
// double-buffered LDS with register staging (1 barrier / tile).
// Qb [BH,S,D] bf16 (pre-scaled), Kb [BH,S,D] bf16, Vt [BH,D,S] bf16.
// Output Ob: merged heads bf16 [B,S,E].
// ---------------------------------------------------------------------------
__global__ __launch_bounds__(256) void attn_kernel(const __bf16* __restrict__ Qb,
                                                   const __bf16* __restrict__ Kb,
                                                   const __bf16* __restrict__ Vt,
                                                   __bf16* __restrict__ Ob) {
  __shared__ __bf16 lK[2][64 * 72];   // [key][d], padded rows (72) vs bank conflicts
  __shared__ __bf16 lV[2][64 * 72];   // [d][key]
  __shared__ __bf16 lP[8 * 16 * 72];  // per-wave P scratch for C->A layout swap

  const int t = threadIdx.x;
  const int w = t >> 5;
  const int lane = t & 31;
  const int ln = lane & 15;
  const int hi = lane >> 4;
  const int bh = blockIdx.y;
  const int qbase = blockIdx.x * 128 + w * 16;
  constexpr int NT = Sc / 64;         // 32 key tiles

  // A-fragments of Q (16x64 = two 16x32 frags), layout per ISA 7.12.2
  FragA aq0, aq1;
  {
    const __bf16* qrow = Qb + ((size_t)bh * Sc + qbase + ln) * Dc;
    aq0.u[0] = *(const uint4*)(qrow + 8 * hi);
    aq0.u[1] = *(const uint4*)(qrow + 16 + 8 * hi);
    aq1.u[0] = *(const uint4*)(qrow + 32 + 8 * hi);
    aq1.u[1] = *(const uint4*)(qrow + 48 + 8 * hi);
  }

  const v8f vzero = {0.f, 0.f, 0.f, 0.f, 0.f, 0.f, 0.f, 0.f};
  v8f acc[4];
  float mrow[8], lrow[8];
  #pragma unroll
  for (int i = 0; i < 4; ++i) acc[i] = vzero;
  #pragma unroll
  for (int r = 0; r < 8; ++r) { mrow[r] = -1e30f; lrow[r] = 0.f; }

  __bf16* lPw = lP + w * (16 * 72);
  const __bf16* Kbh = Kb + (size_t)bh * Sc * Dc;
  const __bf16* Vbh = Vt + (size_t)bh * Dc * Sc;

  // per-thread staging coordinates: two 16B chunks per tile (rows r0, r0+32)
  const int r0 = t >> 3;            // 0..31
  const int o0 = (t & 7) * 8;       // 0..56

  // stage tile 0 into buffer 0
  *(uint4*)&lK[0][(r0)      * 72 + o0] = *(const uint4*)(Kbh + (size_t)(r0)      * Dc + o0);
  *(uint4*)&lK[0][(r0 + 32) * 72 + o0] = *(const uint4*)(Kbh + (size_t)(r0 + 32) * Dc + o0);
  *(uint4*)&lV[0][(r0)      * 72 + o0] = *(const uint4*)(Vbh + (size_t)(r0)      * Sc + o0);
  *(uint4*)&lV[0][(r0 + 32) * 72 + o0] = *(const uint4*)(Vbh + (size_t)(r0 + 32) * Sc + o0);
  __syncthreads();

  for (int kt = 0; kt < NT; ++kt) {
    const int cur = kt & 1, nxt = cur ^ 1;
    const bool have_next = (kt + 1 < NT);

    // issue next tile's global loads now; latency hides behind this tile's math
    uint4 nk0{}, nk1{}, nv0{}, nv1{};
    if (have_next) {
      const __bf16* kg = Kbh + (size_t)((kt + 1) * 64) * Dc;
      const __bf16* vg = Vbh + (kt + 1) * 64;
      nk0 = *(const uint4*)(kg + (size_t)(r0)      * Dc + o0);
      nk1 = *(const uint4*)(kg + (size_t)(r0 + 32) * Dc + o0);
      nv0 = *(const uint4*)(vg + (size_t)(r0)      * Sc + o0);
      nv1 = *(const uint4*)(vg + (size_t)(r0 + 32) * Sc + o0);
      if (kt + 2 < NT) {  // warm L2 two tiles ahead
        __builtin_prefetch(Kbh + (size_t)((kt + 2) * 64 + r0) * Dc + o0, 0, 1);
        __builtin_prefetch(Vbh + (size_t)r0 * Sc + (kt + 2) * 64 + o0, 0, 1);
      }
    }

    // S = Q K^T : 16x64 scores per wave (4 N-tiles x 2 k-steps)
    v8f st[4];
    #pragma unroll
    for (int nt = 0; nt < 4; ++nt) {
      FragA bk0, bk1;
      const __bf16* kr = &lK[cur][(nt * 16 + ln) * 72 + 8 * hi];
      bk0.u[0] = *(const uint4*)(kr);
      bk0.u[1] = *(const uint4*)(kr + 16);
      bk1.u[0] = *(const uint4*)(kr + 32);
      bk1.u[1] = *(const uint4*)(kr + 48);
      v8f s = vzero;
      s = WMMA_BF16(aq0.v, bk0.v, s);
      s = WMMA_BF16(aq1.v, bk1.v, s);
      st[nt] = s;
    }

    // online softmax: 16-lane row reductions via DPP (VALU only, no LDS waits)
    float corr[8];
    #pragma unroll
    for (int r = 0; r < 8; ++r) {
      float mx = fmaxf(fmaxf(st[0][r], st[1][r]), fmaxf(st[2][r], st[3][r]));
      mx = red16_max(mx);
      const float mnew = fmaxf(mrow[r], mx);
      corr[r] = __expf(mrow[r] - mnew);
      mrow[r] = mnew;
    }
    #pragma unroll
    for (int nt = 0; nt < 4; ++nt)
      #pragma unroll
      for (int r = 0; r < 8; ++r)
        st[nt][r] = __expf(st[nt][r] - mrow[r]);
    #pragma unroll
    for (int r = 0; r < 8; ++r) {
      float sum = red16_sum((st[0][r] + st[1][r]) + (st[2][r] + st[3][r]));
      lrow[r] = lrow[r] * corr[r] + sum;
    }
    #pragma unroll
    for (int nt = 0; nt < 4; ++nt)
      #pragma unroll
      for (int r = 0; r < 8; ++r)
        acc[nt][r] *= corr[r];

    // P: C layout -> LDS -> A layout (per-wave region, in-order DS within wave)
    #pragma unroll
    for (int nt = 0; nt < 4; ++nt)
      #pragma unroll
      for (int r = 0; r < 8; ++r)
        lPw[(r + 8 * hi) * 72 + nt * 16 + ln] = (__bf16)st[nt][r];

    FragA ap0, ap1;
    {
      const __bf16* pr = lPw + ln * 72;
      ap0.u[0] = *(const uint4*)(pr + 8 * hi);
      ap0.u[1] = *(const uint4*)(pr + 16 + 8 * hi);
      ap1.u[0] = *(const uint4*)(pr + 32 + 8 * hi);
      ap1.u[1] = *(const uint4*)(pr + 48 + 8 * hi);
    }

    // O += P V  (V stored transposed: lV[d][key], so B frag is contiguous)
    #pragma unroll
    for (int dt = 0; dt < 4; ++dt) {
      FragA bv0, bv1;
      const __bf16* vr = &lV[cur][(dt * 16 + ln) * 72 + 8 * hi];
      bv0.u[0] = *(const uint4*)(vr);
      bv0.u[1] = *(const uint4*)(vr + 16);
      bv1.u[0] = *(const uint4*)(vr + 32);
      bv1.u[1] = *(const uint4*)(vr + 48);
      acc[dt] = WMMA_BF16(ap0.v, bv0.v, acc[dt]);
      acc[dt] = WMMA_BF16(ap1.v, bv1.v, acc[dt]);
    }

    // commit staged registers into the other buffer (safe: last read of that
    // buffer completed before the previous barrier), then one barrier.
    if (have_next) {
      *(uint4*)&lK[nxt][(r0)      * 72 + o0] = nk0;
      *(uint4*)&lK[nxt][(r0 + 32) * 72 + o0] = nk1;
      *(uint4*)&lV[nxt][(r0)      * 72 + o0] = nv0;
      *(uint4*)&lV[nxt][(r0 + 32) * 72 + o0] = nv1;
    }
    __syncthreads();
  }

  // normalize, merge heads, store bf16 [B,S,E]
  const int b = bh >> 4, h = bh & 15;
  float rinv[8];
  #pragma unroll
  for (int r = 0; r < 8; ++r) rinv[r] = 1.0f / lrow[r];
  #pragma unroll
  for (int dt = 0; dt < 4; ++dt)
    #pragma unroll
    for (int r = 0; r < 8; ++r) {
      const int row = qbase + r + 8 * hi;
      Ob[((size_t)(b * Sc) + row) * Ec + h * Dc + dt * 16 + ln] =
          (__bf16)(acc[dt][r] * rinv[r]);
    }
}

// ---------------------------------------------------------------------------
// Phase 2: out = O @ w_out^T + b_out.  M=8192, N=1024, K=1024, bf16 WMMA.
// ---------------------------------------------------------------------------
__global__ __launch_bounds__(256) void proj_kernel(const __bf16* __restrict__ Ob,
                                                   const __bf16* __restrict__ Wb,
                                                   const float* __restrict__ bias,
                                                   float* __restrict__ out) {
  const int t = threadIdx.x, w = t >> 5, lane = t & 31;
  const int ln = lane & 15, hi = lane >> 4;
  const int mbase = blockIdx.x * 128 + w * 16;
  const int nbase = blockIdx.y * 64;
  const v8f vzero = {0.f, 0.f, 0.f, 0.f, 0.f, 0.f, 0.f, 0.f};
  v8f acc[4];
  #pragma unroll
  for (int i = 0; i < 4; ++i) acc[i] = vzero;

  const __bf16* arow = Ob + (size_t)(mbase + ln) * Ec;
  for (int k0 = 0; k0 < Ec; k0 += 32) {
    FragA a;
    a.u[0] = *(const uint4*)(arow + k0 + 8 * hi);
    a.u[1] = *(const uint4*)(arow + k0 + 16 + 8 * hi);
    #pragma unroll
    for (int nt = 0; nt < 4; ++nt) {
      FragA bw;
      const __bf16* wr = Wb + (size_t)(nbase + nt * 16 + ln) * Ec + k0 + 8 * hi;
      bw.u[0] = *(const uint4*)(wr);
      bw.u[1] = *(const uint4*)(wr + 16);
      acc[nt] = WMMA_BF16(a.v, bw.v, acc[nt]);
    }
  }
  #pragma unroll
  for (int nt = 0; nt < 4; ++nt) {
    const float bv = bias[nbase + nt * 16 + ln];
    #pragma unroll
    for (int r = 0; r < 8; ++r) {
      const int row = mbase + r + 8 * hi;
      out[(size_t)row * Ec + nbase + nt * 16 + ln] = acc[nt][r] + bv;
    }
  }
}

// ---------------------------------------------------------------------------
extern "C" void kernel_launch(void* const* d_in, const int* in_sizes, int n_in,
                              void* d_out, int out_size, void* d_ws, size_t ws_size,
                              hipStream_t stream) {
  const float* q     = (const float*)d_in[0];
  const float* k     = (const float*)d_in[1];
  const float* v     = (const float*)d_in[2];
  const float* w_out = (const float*)d_in[3];
  const float* b_out = (const float*)d_in[4];

  __bf16* Qb = (__bf16*)d_ws;          // 16 MB each segment
  __bf16* Kb = Qb + QK_ELEMS;
  __bf16* Vt = Kb + QK_ELEMS;
  __bf16* Ob = Vt + QK_ELEMS;
  __bf16* Wb = Ob + QK_ELEMS;          // 2 MB

  cvt_split_kernel<<<8192, 256, 0, stream>>>(q, Qb, 0.125f);  // scale = D^-0.5 folded
  cvt_split_kernel<<<8192, 256, 0, stream>>>(k, Kb, 1.0f);
  vtrans_kernel<<<dim3(Sc / 32, Dc / 32, Bc * Hc), dim3(32, 8), 0, stream>>>(v, Vt);
  cvt_w_kernel<<<(Ec * Ec / 4) / 256, 256, 0, stream>>>(w_out, Wb);
  attn_kernel<<<dim3(Sc / 128, Bc * Hc), 256, 0, stream>>>(Qb, Kb, Vt, Ob);
  proj_kernel<<<dim3(Bc * Sc / 128, Ec / 64), 256, 0, stream>>>(Ob, Wb, b_out, (float*)d_out);
}
